// VaeCap_25941602468318
// MI455X (gfx1250) — matrix operand order repaired and checked by reading, not compiled, not run on hardware
//
#include <hip/hip_runtime.h>
#include <math.h>

// ---------------------------------------------------------------------------
// CapsNet + VAE head for MI455X (gfx1250, wave32, WMMA).
//
// Heavy math on v_wmma_f32_16x16x32_bf16:
//   - conv3x3 as implicit GEMM (19.3 GFLOP), B tile staged to LDS with
//     GLOBAL_LOAD_ASYNC_TO_LDS_B128 (ASYNCcnt double-buffer), fused
//     BN+ReLU+squash epilogue
//   - routing iter 1 (uniform c) as GEMM over K=I*IDC=32768 (2.1 GFLOP)
// Routing iters 2/3 recompute x_hat from pose (L2-resident caps_w) instead of
// materializing 537 MB of x_hat; only b-logits (33.5 MB) are stored.
// ---------------------------------------------------------------------------

typedef __attribute__((ext_vector_type(16))) __bf16 v16bf;
typedef __attribute__((ext_vector_type(8)))  float  v8f;
typedef __attribute__((ext_vector_type(4)))  int    v4i;

union FragAB { v16bf v; uint4 q[2]; unsigned short u[16]; };
union FragC  { v8f  v; float f[8]; };
union Pack8  { uint4 q; unsigned short u[8]; };

#ifndef __has_builtin
#define __has_builtin(x) 0
#endif

#define GLOBAL_AS __attribute__((address_space(1)))
#define LDS_AS    __attribute__((address_space(3)))

#if __has_builtin(__builtin_amdgcn_global_load_async_to_lds_b128)
#define HAVE_ASYNC_LDS 1
#else
#define HAVE_ASYNC_LDS 0
#endif

// 16-byte global -> LDS copy; async (ASYNCcnt) when available.
// Builtin signature (from hipcc diagnostic): (AS1 v4i*, AS3 v4i*, imm, imm).
__device__ __forceinline__ void cp16_g2l(unsigned short* l, const unsigned short* g) {
#if HAVE_ASYNC_LDS
  __builtin_amdgcn_global_load_async_to_lds_b128(
      (GLOBAL_AS v4i*)g, (LDS_AS v4i*)l, 0, 0);
#else
  *(uint4*)l = *(const uint4*)g;
#endif
}

#if HAVE_ASYNC_LDS
#if __has_builtin(__builtin_amdgcn_s_wait_asynccnt)
#define WAIT_ASYNC(n) __builtin_amdgcn_s_wait_asynccnt(n)
#else
#define WAIT_ASYNC(n) asm volatile("s_wait_asynccnt %0" ::"i"(n) : "memory")
#endif
#else
#define WAIT_ASYNC(n) ((void)0)
#endif

// ---- problem sizes --------------------------------------------------------
#define BB    64
#define PP    512
#define SS    8
#define OO    32
#define ODIM  16
#define IDC   8
#define II    4096          // S*S*(P/IDC)
#define ZZ    64
#define HH    1024
#define KFLAT (II*IDC)      // 32768

// ---- workspace layout (bytes) ---------------------------------------------
#define OFF_XPAD  ((size_t)0)           // bf16 [64][10][10][512]  (ch-last, zero pad)
#define OFF_WB    ((size_t)6553600)     // bf16 [512 co][9 tap][512 ci]
#define OFF_WCAPS ((size_t)11272192)    // bf16 [32 o][16 d][32768 kflat]
#define OFF_POSE  ((size_t)44826624)    // bf16 [64 b][4096 i][8 k]
#define OFF_B1    ((size_t)49020928)    // f32  [64 b][32 o][4096 i]
#define OFF_S1    ((size_t)82575360)    // f32  [64][32][16]
#define OFF_S2    (OFF_S1   + 131072)
#define OFF_SOUT  (OFF_S2   + 131072)
#define OFF_V1    (OFF_SOUT + 131072)
#define OFF_V2    (OFF_V1   + 131072)
#define OFF_OUTV  (OFF_V2   + 131072)   // f32 [64][32][16] squashed routing out

__device__ __forceinline__ unsigned short f2bf(float f) {
  union { float f; unsigned u; } v; v.f = f;
  unsigned r = 0x7FFFu + ((v.u >> 16) & 1u);   // round-to-nearest-even
  return (unsigned short)((v.u + r) >> 16);
}
__device__ __forceinline__ float bf2f(unsigned short h) {
  union { unsigned u; float f; } v; v.u = ((unsigned)h) << 16;
  return v.f;
}

// ---------------------------------------------------------------------------
__global__ __launch_bounds__(256) void k_zero(float* p, int n) {
  int t = blockIdx.x * 256 + threadIdx.x;
  if (t < n) p[t] = 0.0f;
}

// x [B][512][8][8] f32 -> xpad bf16 [B][10][10][512] channel-last, zero border
__global__ __launch_bounds__(256) void k_prep_x(const float* x, unsigned short* xpad) {
  int t = blockIdx.x * 256 + threadIdx.x;
  if (t >= BB * 10 * 10 * PP) return;
  int ci = t & 511; int r = t >> 9;
  int w = r % 10; r /= 10; int h = r % 10; int b = r / 10;
  float v = 0.0f;
  if (h >= 1 && h <= 8 && w >= 1 && w <= 8)
    v = x[(((size_t)b * PP + ci) * SS + (h - 1)) * SS + (w - 1)];
  xpad[t] = f2bf(v);
}

// conv_w [512 co][512 ci][3][3] -> Wb bf16 [co][tap][ci]
__global__ __launch_bounds__(256) void k_prep_w(const float* cw, unsigned short* wb) {
  int t = blockIdx.x * 256 + threadIdx.x;
  if (t >= PP * 9 * PP) return;
  int ci = t & 511; int r = t >> 9;
  int tap = r % 9; int co = r / 9;
  int kh = tap / 3, kw = tap % 3;
  wb[t] = f2bf(cw[(((size_t)co * PP + ci) * 3 + kh) * 3 + kw]);
}

// caps_w f32 [o][i][d][k] -> bf16 [o][d][kflat=i*8+k]  (B-fragment friendly)
__global__ __launch_bounds__(256) void k_prep_cw(const float* cw, unsigned short* wc) {
  int t = blockIdx.x * 256 + threadIdx.x;
  if (t >= OO * ODIM * KFLAT) return;
  int k = t & 7; int r = t >> 3;
  int i = r & 4095; r >>= 12;
  int d = r & 15; int o = r >> 4;
  wc[t] = f2bf(cw[(((size_t)o * II + i) * ODIM + d) * IDC + k]);
}

// ---------------------------------------------------------------------------
// Conv3x3 implicit GEMM: C[co, pix] = sum_{tap,ci} W[co,tap,ci] * Xpad[...,ci]
// block = 128 (4 waves) sharing one B tile (32ci x 64pix bf16, double-buffered
// in LDS via async-to-LDS). Each wave: 16 co x 64 pixels (4 wmma accums).
// grid = (64 batches, 8 co-groups). Epilogue: BN+ReLU+squash -> pose bf16.
// ---------------------------------------------------------------------------
#define BROW 40   // LDS row stride (ushorts): 80B, 16B aligned, conflict-free

__global__ __launch_bounds__(128) void k_conv(
    const unsigned short* __restrict__ wb, const unsigned short* __restrict__ xp,
    const float* __restrict__ gamma, const float* __restrict__ beta,
    const float* __restrict__ mean,  const float* __restrict__ var,
    unsigned short* __restrict__ pose) {
  __shared__ unsigned short bstage[2][64 * BROW];   // [pixel][ci(32)+pad]
  const int tid  = threadIdx.x;
  const int lane = tid & 31;
  const int wave = tid >> 5;
  const int l16  = lane & 15;
  const int hi   = lane >> 4;
  const int b    = blockIdx.x;
  const int co0  = blockIdx.y * 64 + wave * 16;

  // staging map: 128 threads cover 64 pixels x 2 halves of 16 ci (32B each)
  const int sp  = tid & 63;        // pixel 0..63
  const int sh  = tid >> 6;        // ci half 0..1
  const int shh = sp >> 3, swp = sp & 7;

  FragC acc[4];
#pragma unroll
  for (int nt = 0; nt < 4; ++nt)
#pragma unroll
    for (int r = 0; r < 8; ++r) acc[nt].f[r] = 0.0f;

  // K-steps: s = t*16 + cc  (tap t in 0..8, ci chunk cc in 0..15) -> 144 steps
  auto issue = [&](int s) {
    const int t = s >> 4, cc = s & 15;
    const int kh = t / 3, kw = t % 3;
    const unsigned short* g =
        xp + (((size_t)(b * 10 + shh + kh)) * 10 + (swp + kw)) * PP + cc * 32 + sh * 16;
    unsigned short* l = &bstage[s & 1][sp * BROW + sh * 16];
    cp16_g2l(l, g);
    cp16_g2l(l + 8, g + 8);
  };

  issue(0);
  for (int s = 0; s < 144; ++s) {
    if (s + 1 < 144) { issue(s + 1); WAIT_ASYNC(2); }
    else             { WAIT_ASYNC(0); }
    __syncthreads();   // all waves' async stripes visible

    const int t = s >> 4, ci0 = (s & 15) * 32;
    // A fragment (16x32 bf16) direct from global (per-wave co tile):
    // lane<16 -> K {0..15}, lane>=16 -> {16..31}
    const unsigned short* ap = wb + ((size_t)(co0 + l16) * 9 + t) * PP + ci0 + hi * 16;
    FragAB a;
    a.q[0] = *(const uint4*)(ap);
    a.q[1] = *(const uint4*)(ap + 8);
    __builtin_prefetch(ap + 32, 0, 0);   // global_prefetch next ci chunk
#pragma unroll
    for (int nt = 0; nt < 4; ++nt) {
      // B fragment (32x16 bf16) from LDS: N=lane&15 (pixel), contiguous K in ci
      const unsigned short* bp = &bstage[s & 1][(nt * 16 + l16) * BROW + hi * 16];
      FragAB bf;
      bf.q[0] = *(const uint4*)(bp);
      bf.q[1] = *(const uint4*)(bp + 8);
      acc[nt].v = __builtin_amdgcn_wmma_f32_16x16x32_bf16(
          false, a.v, false, bf.v, (short)0, acc[nt].v, false, false);
    }
    __syncthreads();   // buffer s&1 free for reuse at step s+2
  }

  // Epilogue: each lane holds 8 consecutive channels = one capsule group.
  float scale[8], shift[8];
#pragma unroll
  for (int r = 0; r < 8; ++r) {
    const int c = co0 + hi * 8 + r;
    const float inv = gamma[c] * rsqrtf(var[c] + 1e-5f);
    scale[r] = inv;
    shift[r] = beta[c] - mean[c] * inv;
  }
  const int g = (co0 >> 3) + hi;   // capsule group id
#pragma unroll
  for (int nt = 0; nt < 4; ++nt) {
    float y[8], msq = 0.0f;
#pragma unroll
    for (int r = 0; r < 8; ++r) {
      y[r] = fmaxf(fmaf(acc[nt].f[r], scale[r], shift[r]), 0.0f);
      msq = fmaf(y[r], y[r], msq);
    }
    const float mag = sqrtf(msq);
    const float sc = msq / (1.0f + msq) / (mag + 1e-8f);
    const int p = nt * 16 + l16;
    const int i = g * 64 + p;
    Pack8 o8;
#pragma unroll
    for (int r = 0; r < 8; ++r) o8.u[r] = f2bf(y[r] * sc);
    *(uint4*)(pose + ((size_t)b * II + i) * IDC) = o8.q;
  }
}

// ---------------------------------------------------------------------------
// Routing iter 1 (c uniform): s1[b,o,d] = sum_kflat pose[b,kflat]*Wc[o,d,kflat]
// Per-o GEMM M=64(b) N=16(d) K=32768, K-split 8, f32 atomics into s1.
// grid = (8 kslice, 4 btile, 32 o), block = 32 (one wave; EXEC all ones).
// ---------------------------------------------------------------------------
__global__ __launch_bounds__(32) void k_r1(
    const unsigned short* __restrict__ pose, const unsigned short* __restrict__ wc,
    float* __restrict__ s1) {
  const int lane = threadIdx.x;
  const int l16 = lane & 15;
  const int hi  = lane >> 4;
  const int b0  = blockIdx.y * 16;
  const int o   = blockIdx.z;
  const int kb0 = blockIdx.x * (KFLAT / 8);

  FragC acc;
#pragma unroll
  for (int r = 0; r < 8; ++r) acc.f[r] = 0.0f;

  const unsigned short* arow = pose + (size_t)(b0 + l16) * KFLAT;
  const unsigned short* brow = wc + ((size_t)o * ODIM + l16) * KFLAT;

  for (int s = 0; s < 128; ++s) {
    const int ka = kb0 + s * 32;
    FragAB a, bf;
    a.q[0]  = *(const uint4*)(arow + ka + hi * 16);
    a.q[1]  = *(const uint4*)(arow + ka + hi * 16 + 8);
    bf.q[0] = *(const uint4*)(brow + ka + hi * 16);
    bf.q[1] = *(const uint4*)(brow + ka + hi * 16 + 8);
    __builtin_prefetch(brow + ka + 32, 0, 0);
    acc.v = __builtin_amdgcn_wmma_f32_16x16x32_bf16(
        false, a.v, false, bf.v, (short)0, acc.v, false, false);
  }
#pragma unroll
  for (int r = 0; r < 8; ++r) {
    const int bb = b0 + hi * 8 + r;
    atomicAdd(&s1[((size_t)bb * OO + o) * ODIM + l16], acc.f[r]);
  }
}

// v = squash(scale * s), one thread per (b,o) row of 16
__global__ __launch_bounds__(256) void k_squash(const float* __restrict__ s,
                                                float* __restrict__ v, float scl) {
  const int row = blockIdx.x * 256 + threadIdx.x;
  if (row >= BB * OO) return;
  float x[16], msq = 0.0f;
#pragma unroll
  for (int d = 0; d < 16; ++d) {
    x[d] = s[(size_t)row * 16 + d] * scl;
    msq = fmaf(x[d], x[d], msq);
  }
  const float mag = sqrtf(msq);
  const float sc = msq / (1.0f + msq) / (mag + 1e-8f);
#pragma unroll
  for (int d = 0; d < 16; ++d) v[(size_t)row * 16 + d] = x[d] * sc;
}

// ---------------------------------------------------------------------------
// Routing iters 2/3: per (b,i): recompute x_hat[o][d] from pose & caps_w (L2),
// logits -> softmax over o -> weighted accumulate (LDS ds_add_f32 + global atomics).
// PASS==2: b1 = v1.x_hat (stored).  PASS==3: logits = b1 + v2.x_hat.
// ---------------------------------------------------------------------------
template <int PASS>
__global__ __launch_bounds__(256) void k_route(
    const unsigned short* __restrict__ pose, const float* __restrict__ caps_w,
    const float* __restrict__ vvec, float* __restrict__ bl, float* __restrict__ sacc) {
  __shared__ float vs[OO * ODIM];
  __shared__ float sl[OO * ODIM];
  const int b = blockIdx.y;
  const int i = blockIdx.x * 256 + threadIdx.x;
  for (int t = threadIdx.x; t < OO * ODIM; t += 256) {
    vs[t] = vvec[(size_t)b * OO * ODIM + t];
    sl[t] = 0.0f;
  }
  __syncthreads();

  float pk[8];
  {
    Pack8 ld;
    ld.q = *(const uint4*)(pose + ((size_t)b * II + i) * IDC);
#pragma unroll
    for (int k = 0; k < 8; ++k) pk[k] = bf2f(ld.u[k]);
  }

  float logit[OO];
  for (int o = 0; o < OO; ++o) {
    const float* wo = caps_w + ((size_t)o * II + i) * (ODIM * IDC);
    float dot = 0.0f;
#pragma unroll
    for (int d = 0; d < ODIM; ++d) {
      float xh = 0.0f;
#pragma unroll
      for (int k = 0; k < 8; ++k) xh = fmaf(wo[d * 8 + k], pk[k], xh);
      dot = fmaf(vs[o * ODIM + d], xh, dot);
    }
    if (PASS == 2) {
      bl[((size_t)b * OO + o) * II + i] = dot;
      logit[o] = dot;
    } else {
      logit[o] = bl[((size_t)b * OO + o) * II + i] + dot;
    }
  }
  float m = logit[0];
  for (int o = 1; o < OO; ++o) m = fmaxf(m, logit[o]);
  float sum = 0.0f;
  for (int o = 0; o < OO; ++o) { logit[o] = __expf(logit[o] - m); sum += logit[o]; }
  const float rs = 1.0f / sum;

  for (int o = 0; o < OO; ++o) {
    const float c = logit[o] * rs;
    const float* wo = caps_w + ((size_t)o * II + i) * (ODIM * IDC);
#pragma unroll
    for (int d = 0; d < ODIM; ++d) {
      float xh = 0.0f;
#pragma unroll
      for (int k = 0; k < 8; ++k) xh = fmaf(wo[d * 8 + k], pk[k], xh);
      atomicAdd(&sl[o * ODIM + d], c * xh);
    }
  }
  __syncthreads();
  for (int t = threadIdx.x; t < OO * ODIM; t += 256)
    atomicAdd(&sacc[(size_t)b * OO * ODIM + t], sl[t]);
}

// ---------------------------------------------------------------------------
// VAE head: per row (b,o): h = relu(out@W1+b1); z = h@W2+b2; zvar = softplus(out@Wv+bv)+1e-8
// ---------------------------------------------------------------------------
__global__ __launch_bounds__(256) void k_head(
    const float* __restrict__ outv, const float* __restrict__ w1,
    const float* __restrict__ b1h, const float* __restrict__ w2,
    const float* __restrict__ b2h, const float* __restrict__ wv,
    const float* __restrict__ bvh, float* __restrict__ dout) {
  __shared__ float ov[ODIM];
  __shared__ float h[HH];
  const int row = blockIdx.x;  // b*32 + o
  if (threadIdx.x < ODIM) ov[threadIdx.x] = outv[(size_t)row * ODIM + threadIdx.x];
  __syncthreads();
  for (int j = threadIdx.x; j < HH; j += 256) {
    float a = b1h[j];
#pragma unroll
    for (int k = 0; k < ODIM; ++k) a = fmaf(ov[k], w1[(size_t)k * HH + j], a);
    h[j] = fmaxf(a, 0.0f);
  }
  __syncthreads();
  if (threadIdx.x < ZZ) {
    const int j = threadIdx.x;
    float z = b2h[j];
    for (int k = 0; k < HH; ++k) z = fmaf(h[k], w2[(size_t)k * ZZ + j], z);
    float a = bvh[j];
#pragma unroll
    for (int k = 0; k < ODIM; ++k) a = fmaf(ov[k], wv[(size_t)k * ZZ + j], a);
    const float sp = (a > 20.0f) ? a : log1pf(__expf(a));
    const float zv = sp + 1e-8f;
    const size_t base = (size_t)row * ZZ + j;
    dout[base] = z;                         // z  (= z_mu in eval)
    dout[131072 + base] = z;                // z_mu
    dout[262144 + base] = zv;               // z_var
  }
}

// ---------------------------------------------------------------------------
extern "C" void kernel_launch(void* const* d_in, const int* in_sizes, int n_in,
                              void* d_out, int out_size, void* d_ws, size_t ws_size,
                              hipStream_t stream) {
  (void)in_sizes; (void)n_in; (void)out_size; (void)ws_size;
  const float* x        = (const float*)d_in[0];
  const float* conv_w   = (const float*)d_in[1];
  const float* bn_gamma = (const float*)d_in[2];
  const float* bn_beta  = (const float*)d_in[3];
  const float* bn_mean  = (const float*)d_in[4];
  const float* bn_var   = (const float*)d_in[5];
  const float* caps_w   = (const float*)d_in[6];
  const float* fcm_w1   = (const float*)d_in[7];
  const float* fcm_b1   = (const float*)d_in[8];
  const float* fcm_w2   = (const float*)d_in[9];
  const float* fcm_b2   = (const float*)d_in[10];
  const float* fcv_w    = (const float*)d_in[11];
  const float* fcv_b    = (const float*)d_in[12];
  float* out = (float*)d_out;

  unsigned char* ws = (unsigned char*)d_ws;
  unsigned short* xpad = (unsigned short*)(ws + OFF_XPAD);
  unsigned short* wb   = (unsigned short*)(ws + OFF_WB);
  unsigned short* wc   = (unsigned short*)(ws + OFF_WCAPS);
  unsigned short* pose = (unsigned short*)(ws + OFF_POSE);
  float* b1    = (float*)(ws + OFF_B1);
  float* s1    = (float*)(ws + OFF_S1);
  float* s2    = (float*)(ws + OFF_S2);
  float* sout  = (float*)(ws + OFF_SOUT);
  float* v1    = (float*)(ws + OFF_V1);
  float* v2    = (float*)(ws + OFF_V2);
  float* outv  = (float*)(ws + OFF_OUTV);

  // zero s1/s2/sout (contiguous 3*32768 floats)
  k_zero<<<(3 * BB * OO * ODIM + 255) / 256, 256, 0, stream>>>(s1, 3 * BB * OO * ODIM);

  k_prep_x<<<(BB * 10 * 10 * PP + 255) / 256, 256, 0, stream>>>(x, xpad);
  k_prep_w<<<(PP * 9 * PP + 255) / 256, 256, 0, stream>>>(conv_w, wb);
  k_prep_cw<<<(OO * ODIM * KFLAT + 255) / 256, 256, 0, stream>>>(caps_w, wc);

  k_conv<<<dim3(BB, 8), 128, 0, stream>>>(wb, xpad, bn_gamma, bn_beta, bn_mean,
                                          bn_var, pose);

  k_r1<<<dim3(8, 4, OO), 32, 0, stream>>>(pose, wc, s1);
  k_squash<<<(BB * OO + 255) / 256, 256, 0, stream>>>(s1, v1, 1.0f / 32.0f);

  k_route<2><<<dim3(II / 256, BB), 256, 0, stream>>>(pose, caps_w, v1, b1, s2);
  k_squash<<<(BB * OO + 255) / 256, 256, 0, stream>>>(s2, v2, 1.0f);

  k_route<3><<<dim3(II / 256, BB), 256, 0, stream>>>(pose, caps_w, v2, b1, sout);
  k_squash<<<(BB * OO + 255) / 256, 256, 0, stream>>>(sout, outv, 1.0f);

  k_head<<<BB * OO, 256, 0, stream>>>(outv, fcm_w1, fcm_b1, fcm_w2, fcm_b2,
                                      fcv_w, fcv_b, out);
}
